// WanSelfAttention_9783935501012
// MI455X (gfx1250) — compile-verified, matrix operand order checked by reading, not run on hardware
//
#include <hip/hip_runtime.h>
#include <hip/hip_bf16.h>

// ---------------------------------------------------------------------------
// WanSelfAttention for MI455X (gfx1250, wave32, WMMA + Tensor Data Mover)
//   DIM=1536, HEADS=12, HEAD_DIM=128, S=4096, B=1
// Pipeline:
//   1) f32->f16 convert of x and Wq/Wk/Wv/Wo
//   2) WMMA GEMM  raw = x @ W^T + b    (TDM double-buffered LDS tiles,
//      TDM row padding for bank-conflict-free ds_load_b128 fragments)
//   3) rmsnorm+rope -> Qh/Kh f16 [h][s][128] ; V -> Vt f16 [h][128][s]
//   4) flash-attention, TDM-staged K/V blocks, v_wmma_f32_16x16x32_f16
//   5) WMMA GEMM  out = attn @ Wo^T + bo  (f32 to d_out)
// ---------------------------------------------------------------------------

#define S_LEN 4096
#define DIM   1536
#define HEADS 12
#define HDIM  128

// padded LDS row strides (halves): +4 DWORDs per row via TDM pad
#define ALD 40    // 32-half rows  -> 40  (20-bank stride, conflict-free)
#define KLD 136   // 128-half rows -> 136 (4-bank stride, conflict-free)
#define PLD 40    // pbuf rows padded the same way

typedef __attribute__((ext_vector_type(16))) _Float16     v16h;
typedef __attribute__((ext_vector_type(8)))  float        v8f;
typedef __attribute__((ext_vector_type(4)))  unsigned int u32x4;
typedef __attribute__((ext_vector_type(8)))  int          i32x8;
typedef __attribute__((ext_vector_type(4)))  int          i32x4;

__device__ __forceinline__ v8f zero8() {
  v8f z = {0.f, 0.f, 0.f, 0.f, 0.f, 0.f, 0.f, 0.f};
  return z;
}

__device__ __forceinline__ v8f wmma_f16(v16h a, v16h b, v8f c) {
  // D = A(16x32 f16) * B(32x16 f16) + C(16x16 f32)
  return __builtin_amdgcn_wmma_f32_16x16x32_f16(
      false, a, false, b, (short)0, c, false, false);
}

// LDS byte offset of a generic pointer into __shared__ (flat addr low 32 bits).
__device__ __forceinline__ unsigned lds_off(const void* p) {
  return (unsigned)(size_t)p;
}

// ---------------------------------------------------------------------------
// Tensor Data Mover: async 2D f16 tile (rows x rowlen elems, row stride in
// elems) global -> LDS. D# per ISA 8.3/8.4: count=1, type=2, data_size=2B,
// tensor dims == tile dims (no OOB clipping). Optional LDS row padding:
// pad_ivl = interval code (2^code qwords per row), pad_amt code = DWORDs-1.
// ---------------------------------------------------------------------------
__device__ __forceinline__ void tdm_load_tile(const _Float16* gptr,
                                              unsigned lds_byte_addr, int rows,
                                              int rowlen, int stride_elems,
                                              int pad_ivl, int pad_amt) {
  const unsigned long long ga = (unsigned long long)(size_t)gptr;
  u32x4 g0;
  g0[0] = 1u;                                  // count=1 (valid), user mode
  g0[1] = lds_byte_addr;                       // lds_addr
  g0[2] = (unsigned)(ga & 0xffffffffull);      // global_addr[31:0]
  g0[3] = (unsigned)((ga >> 32) & 0x1ffffffu)  // global_addr[56:32]
          | 0x80000000u;                       // type=2 ("image")
  i32x8 g1;
  g1[0] = (1 << 16)                            // data_size=1 -> 2 bytes
          | (1 << 20)                          // pad_enable
          | (pad_ivl << 22)                    // pad_interval
          | (pad_amt << 25);                   // pad_amount
  g1[1] = (rowlen & 0xffff) << 16;             // tensor_dim0[15:0]
  g1[2] = (rows & 0xffff) << 16;               // td0 hi=0 | tensor_dim1 lo
  g1[3] = (rowlen & 0xffff) << 16;             // td1 hi=0 | tile_dim0
  g1[4] = rows & 0xffff;                       // tile_dim1 | tile_dim2=0
  g1[5] = stride_elems;                        // tensor_dim0_stride[31:0]
  g1[6] = 0;                                   // stride hi | dim1_stride lo
  g1[7] = 0;
  i32x4 z4 = {0, 0, 0, 0};
#if __clang_major__ >= 23
  i32x8 z8 = {0, 0, 0, 0, 0, 0, 0, 0};
  __builtin_amdgcn_tensor_load_to_lds(g0, g1, z4, z4, z8, 0);
#else
  __builtin_amdgcn_tensor_load_to_lds(g0, g1, z4, z4, 0);
#endif
}

// Load a 16x32 f16 WMMA A/B fragment from a row-major tile (global or LDS):
//   lanes 0-15 : halves[0..7]=K(k0..k0+7),  halves[8..15]=K(k0+16..k0+23)
//   lanes16-31 : halves[0..7]=K(k0+8..+15), halves[8..15]=K(k0+24..+31)
__device__ __forceinline__ v16h load_frag(const _Float16* __restrict__ base,
                                          int ld, int row0, int k0, int lane) {
  const _Float16* p =
      base + (size_t)(row0 + (lane & 15)) * ld + k0 + ((lane >> 4) << 3);
  v16h r;
  ((uint4*)&r)[0] = *(const uint4*)p;
  ((uint4*)&r)[1] = *(const uint4*)(p + 16);
  return r;
}

// ---------------------------------------------------------------------------
// f32 -> f16 elementwise convert
// ---------------------------------------------------------------------------
__global__ void cvt_f16_kernel(const float* __restrict__ in,
                               _Float16* __restrict__ out, int n) {
  int i = blockIdx.x * 256 + threadIdx.x;
  if (i < n) out[i] = (_Float16)in[i];
}

// ---------------------------------------------------------------------------
// GEMM: C[M x N] = A[M x K](f16) * B[N x K](f16)^T + bias[N], f32 output.
// M=4096, N=K=1536. Block = 256 thr (8 waves); block tile 128x256; wave tile
// 64x64. A(128x32) and B(256x32) k-slabs stream through double-buffered,
// row-padded LDS via TDM, overlapped with WMMA using TENSORcnt.
// ---------------------------------------------------------------------------
__global__ __launch_bounds__(256) void gemm_wmma_kernel(
    const _Float16* __restrict__ A, const _Float16* __restrict__ B,
    const float* __restrict__ bias, float* __restrict__ C) {
  const int N_ = DIM, K_ = DIM;
  const int KSTEPS = K_ / 32;  // 48
  __shared__ _Float16 Asm[2][128 * ALD];  // 2 x 10 KB (padded rows)
  __shared__ _Float16 Bsm[2][256 * ALD];  // 2 x 20 KB (padded rows)
  const int lane = threadIdx.x & 31;
  const int wid = threadIdx.x >> 5;
  const int nb = N_ / 256;  // 6
  const int bm = blockIdx.x / nb;
  const int bn = blockIdx.x % nb;
  const _Float16* Ablk = A + (size_t)bm * 128 * K_;
  const _Float16* Bblk = B + (size_t)bn * 256 * K_;
  const int mloc = (wid & 1) * 64;
  const int nloc = (wid >> 1) * 64;

  v8f acc[4][4];
#pragma unroll
  for (int i = 0; i < 4; i++)
#pragma unroll
    for (int j = 0; j < 4; j++) acc[i][j] = zero8();

  if (wid == 0) {  // prime buffer 0
    tdm_load_tile(Ablk, lds_off(&Asm[0][0]), 128, 32, K_, 3, 3);
    tdm_load_tile(Bblk, lds_off(&Bsm[0][0]), 256, 32, K_, 3, 3);
  }
  for (int ks = 0; ks < KSTEPS; ks++) {
    const int cur = ks & 1;
    if (wid == 0) {
      if (ks + 1 < KSTEPS) {  // issue next slab, retire previous pair
        tdm_load_tile(Ablk + (ks + 1) * 32, lds_off(&Asm[cur ^ 1][0]), 128, 32, K_, 3, 3);
        tdm_load_tile(Bblk + (ks + 1) * 32, lds_off(&Bsm[cur ^ 1][0]), 256, 32, K_, 3, 3);
        __builtin_amdgcn_s_wait_tensorcnt(2);
      } else {
        __builtin_amdgcn_s_wait_tensorcnt(0);
      }
    }
    __syncthreads();  // buf[cur] ready for everyone
    v16h a[4], b[4];
#pragma unroll
    for (int i = 0; i < 4; i++) a[i] = load_frag(&Asm[cur][0], ALD, mloc + 16 * i, 0, lane);
#pragma unroll
    for (int j = 0; j < 4; j++) b[j] = load_frag(&Bsm[cur][0], ALD, nloc + 16 * j, 0, lane);
#pragma unroll
    for (int i = 0; i < 4; i++)
#pragma unroll
      for (int j = 0; j < 4; j++) acc[i][j] = wmma_f16(a[i], b[j], acc[i][j]);
    __syncthreads();  // all reads of buf[cur] done before it is refilled
  }

  const int m0 = bm * 128 + mloc;
  const int n0 = bn * 256 + nloc;
  const int rhi = (lane >> 4) << 3;  // +8 rows for lanes 16..31
#pragma unroll
  for (int i = 0; i < 4; i++) {
#pragma unroll
    for (int j = 0; j < 4; j++) {
      const int col = n0 + 16 * j + (lane & 15);
      const float bv = bias[col];
#pragma unroll
      for (int r = 0; r < 8; r++) {
        const int row = m0 + 16 * i + r + rhi;
        C[(size_t)row * N_ + col] = acc[i][j][r] + bv;
      }
    }
  }
}

// ---------------------------------------------------------------------------
// RMSNorm (over full 1536) + gain + RoPE, f32 raw -> f16 out [h][s][128].
// ---------------------------------------------------------------------------
__global__ __launch_bounds__(256) void norm_rope_kernel(
    const float* __restrict__ raw, const float* __restrict__ g,
    const float* __restrict__ freqs, _Float16* __restrict__ out) {
  const int s = blockIdx.x;
  const int t = threadIdx.x;
  const int lane = t & 31, wid = t >> 5;
  const float* row = raw + (size_t)s * DIM;

  float ss = 0.f;
  for (int i = t; i < DIM; i += 256) {
    float v = row[i];
    ss += v * v;
  }
#pragma unroll
  for (int off = 16; off; off >>= 1) ss += __shfl_xor(ss, off, 32);
  __shared__ float red[8];
  if (lane == 0) red[wid] = ss;
  __syncthreads();
  float tot = 0.f;
#pragma unroll
  for (int w = 0; w < 8; w++) tot += red[w];
  const float inv = rsqrtf(tot / (float)DIM + 1e-6f);

  for (int p = t; p < DIM / 2; p += 256) {
    const int i0 = 2 * p;
    const float y0 = row[i0] * inv * g[i0];
    const float y1 = row[i0 + 1] * inv * g[i0 + 1];
    const int h = i0 >> 7;
    const int d = i0 & 127;
    const float ang = freqs[(size_t)s * 64 + (d >> 1)];
    float sn, cs;
    __sincosf(ang, &sn, &cs);
    _Float16* dst = out + ((size_t)h * S_LEN + s) * HDIM + d;
    dst[0] = (_Float16)(y0 * cs - y1 * sn);
    dst[1] = (_Float16)(y0 * sn + y1 * cs);
  }
}

// ---------------------------------------------------------------------------
// V raw f32 [s][1536] -> Vt f16 [h][128][s]
// ---------------------------------------------------------------------------
__global__ void v_transpose_kernel(const float* __restrict__ raw,
                                   _Float16* __restrict__ Vt) {
  int i = blockIdx.x * 256 + threadIdx.x;
  if (i >= S_LEN * DIM) return;
  const int s = i / DIM;
  const int c = i % DIM;
  const int h = c >> 7, d = c & 127;
  Vt[((size_t)(h * HDIM + d)) * S_LEN + s] = (_Float16)raw[i];
}

// ---------------------------------------------------------------------------
// Flash attention. 8 waves/block, one wave per 16-row Q tile; all waves of a
// block share the same head, so 32-key K / Vt slabs are TDM-streamed once per
// block into double-buffered, row-padded LDS. Online softmax, 32-key blocks.
// grid = HEADS * (S/16) / 8 = 384.
// ---------------------------------------------------------------------------
__global__ __launch_bounds__(256) void attn_kernel(
    const _Float16* __restrict__ Qh, const _Float16* __restrict__ Kh,
    const _Float16* __restrict__ Vt, _Float16* __restrict__ Oo) {
  __shared__ _Float16 Ksm[2][32 * KLD];   // [key][dim], padded rows
  __shared__ _Float16 Vsm[2][128 * ALD];  // [dim][key], padded rows
  __shared__ _Float16 pbuf[8][16 * PLD];  // per-wave P staging (C->A relayout)
  const int lane = threadIdx.x & 31;
  const int wid = threadIdx.x >> 5;
  const int gw = blockIdx.x * 8 + wid;
  const int h = gw >> 8;  // 256 Q tiles per head; 8 per block -> uniform head
  const int q0 = (gw & 255) * 16;
  const _Float16* Q = Qh + (size_t)h * S_LEN * HDIM;
  const _Float16* K = Kh + (size_t)h * S_LEN * HDIM;
  const _Float16* V = Vt + (size_t)h * HDIM * S_LEN;  // [128][S]
  const float scale = 0.08838834764831845f;           // 1/sqrt(128)
  const int rhi = (lane >> 4) << 3;

  v16h aq[4];
#pragma unroll
  for (int k = 0; k < 4; k++) aq[k] = load_frag(Q, HDIM, q0, 32 * k, lane);

  v8f o[8];
#pragma unroll
  for (int t = 0; t < 8; t++) o[t] = zero8();
  float mrow[8], lrow[8];
#pragma unroll
  for (int r = 0; r < 8; r++) {
    mrow[r] = -1e30f;
    lrow[r] = 0.f;
  }

  _Float16* pw = &pbuf[wid][0];

  if (wid == 0) {  // prime buffer 0 with first 32-key slab
    tdm_load_tile(K, lds_off(&Ksm[0][0]), 32, 128, 128, 5, 3);
    tdm_load_tile(V, lds_off(&Vsm[0][0]), 128, 32, S_LEN, 3, 3);
  }
  for (int jb = 0; jb < S_LEN; jb += 32) {
    const int cur = (jb >> 5) & 1;
    if (wid == 0) {
      if (jb + 32 < S_LEN) {
        tdm_load_tile(K + (size_t)(jb + 32) * HDIM, lds_off(&Ksm[cur ^ 1][0]), 32, 128, 128, 5, 3);
        tdm_load_tile(V + (jb + 32), lds_off(&Vsm[cur ^ 1][0]), 128, 32, S_LEN, 3, 3);
        __builtin_amdgcn_s_wait_tensorcnt(2);
      } else {
        __builtin_amdgcn_s_wait_tensorcnt(0);
      }
    }
    __syncthreads();  // K/V slab [cur] visible to all waves
    // ---- S = Q K^T for this 32-key slab (two 16-wide N tiles) ----
    v8f s0 = zero8(), s1 = zero8();
#pragma unroll
    for (int k = 0; k < 4; k++) {
      v16h b0 = load_frag(&Ksm[cur][0], KLD, 0, 32 * k, lane);
      v16h b1 = load_frag(&Ksm[cur][0], KLD, 16, 32 * k, lane);
      s0 = wmma_f16(aq[k], b0, s0);
      s1 = wmma_f16(aq[k], b1, s1);
    }
    // ---- V fragments into registers (all LDS slab reads before barrier) ----
    v16h vf[8];
#pragma unroll
    for (int t = 0; t < 8; t++) vf[t] = load_frag(&Vsm[cur][0], ALD, 16 * t, 0, lane);
    // ---- online softmax; write P to per-wave LDS in [row][key] order ----
    float csc[8];
#pragma unroll
    for (int r = 0; r < 8; r++) {
      const float v0 = s0[r] * scale;
      const float v1 = s1[r] * scale;
      float mx = fmaxf(v0, v1);
#pragma unroll
      for (int off = 1; off < 16; off <<= 1) mx = fmaxf(mx, __shfl_xor(mx, off, 32));
      const float mn = fmaxf(mrow[r], mx);
      csc[r] = __expf(mrow[r] - mn);
      mrow[r] = mn;
      const float p0 = __expf(v0 - mn);
      const float p1 = __expf(v1 - mn);
      float sum = p0 + p1;
#pragma unroll
      for (int off = 1; off < 16; off <<= 1) sum += __shfl_xor(sum, off, 32);
      lrow[r] = lrow[r] * csc[r] + sum;
      const int prow = r + rhi;
      pw[prow * PLD + (lane & 15)] = (_Float16)p0;
      pw[prow * PLD + 16 + (lane & 15)] = (_Float16)p1;
    }
    __syncthreads();  // pbuf ready; all slab reads retired before refill
    // ---- reload P in A-fragment layout ----
    v16h ap;
    {
      const _Float16* pr = pw + (lane & 15) * PLD + ((lane >> 4) << 3);
      ((uint4*)&ap)[0] = *(const uint4*)pr;
      ((uint4*)&ap)[1] = *(const uint4*)(pr + 16);
    }
    // ---- O = O*c + P V ----
#pragma unroll
    for (int t = 0; t < 8; t++) {
#pragma unroll
      for (int r = 0; r < 8; r++) o[t][r] *= csc[r];
      o[t] = wmma_f16(ap, vf[t], o[t]);
    }
  }

  // ---- normalize and store f16 [s][h*128+d] ----
#pragma unroll
  for (int r = 0; r < 8; r++) lrow[r] = 1.f / lrow[r];
#pragma unroll
  for (int t = 0; t < 8; t++) {
#pragma unroll
    for (int r = 0; r < 8; r++) {
      const int row = q0 + r + rhi;
      const int col = h * HDIM + 16 * t + (lane & 15);
      Oo[(size_t)row * DIM + col] = (_Float16)(o[t][r] * lrow[r]);
    }
  }
}

// ---------------------------------------------------------------------------
// Host launcher
// ---------------------------------------------------------------------------
extern "C" void kernel_launch(void* const* d_in, const int* in_sizes, int n_in,
                              void* d_out, int out_size, void* d_ws,
                              size_t ws_size, hipStream_t stream) {
  (void)in_sizes; (void)n_in; (void)out_size; (void)ws_size;
  const float* x  = (const float*)d_in[0];
  const float* fr = (const float*)d_in[1];
  const float* Wq = (const float*)d_in[2];
  const float* bq = (const float*)d_in[3];
  const float* Wk = (const float*)d_in[4];
  const float* bk = (const float*)d_in[5];
  const float* Wv = (const float*)d_in[6];
  const float* bv = (const float*)d_in[7];
  const float* Wo = (const float*)d_in[8];
  const float* bo = (const float*)d_in[9];
  const float* gq = (const float*)d_in[10];
  const float* gk = (const float*)d_in[11];
  float* out = (float*)d_out;

  char* ws = (char*)d_ws;
  const size_t SD_H = (size_t)S_LEN * DIM * 2;   // f16 S*DIM   = 12,582,912 B
  const size_t WW_H = (size_t)DIM * DIM * 2;     // f16 DIM*DIM =  4,718,592 B
  _Float16* xh   = (_Float16*)(ws);
  _Float16* Wqh  = (_Float16*)(ws + SD_H);
  _Float16* Wkh  = (_Float16*)(ws + SD_H + WW_H);
  _Float16* Wvh  = (_Float16*)(ws + SD_H + 2 * WW_H);
  _Float16* Woh  = (_Float16*)(ws + SD_H + 3 * WW_H);
  _Float16* Qh   = (_Float16*)(ws + SD_H + 4 * WW_H);
  _Float16* Kh   = (_Float16*)(ws + 2 * SD_H + 4 * WW_H);
  _Float16* Vth  = (_Float16*)(ws + 3 * SD_H + 4 * WW_H);
  _Float16* attn = (_Float16*)(ws + 4 * SD_H + 4 * WW_H);
  float*    raw  = (float*)   (ws + 5 * SD_H + 4 * WW_H);  // f32 S*DIM

  const int nSD = S_LEN * DIM;   // 6,291,456
  const int nWW = DIM * DIM;     // 2,359,296
  cvt_f16_kernel<<<(nSD + 255) / 256, 256, 0, stream>>>(x, xh, nSD);
  cvt_f16_kernel<<<(nWW + 255) / 256, 256, 0, stream>>>(Wq, Wqh, nWW);
  cvt_f16_kernel<<<(nWW + 255) / 256, 256, 0, stream>>>(Wk, Wkh, nWW);
  cvt_f16_kernel<<<(nWW + 255) / 256, 256, 0, stream>>>(Wv, Wvh, nWW);
  cvt_f16_kernel<<<(nWW + 255) / 256, 256, 0, stream>>>(Wo, Woh, nWW);

  const int GEMM_GRID = (S_LEN / 128) * (DIM / 256);  // 192

  // Q projection -> rmsnorm+rope -> Qh
  gemm_wmma_kernel<<<GEMM_GRID, 256, 0, stream>>>(xh, Wqh, bq, raw);
  norm_rope_kernel<<<S_LEN, 256, 0, stream>>>(raw, gq, fr, Qh);
  // K projection -> rmsnorm+rope -> Kh
  gemm_wmma_kernel<<<GEMM_GRID, 256, 0, stream>>>(xh, Wkh, bk, raw);
  norm_rope_kernel<<<S_LEN, 256, 0, stream>>>(raw, gk, fr, Kh);
  // V projection -> transpose -> Vth
  gemm_wmma_kernel<<<GEMM_GRID, 256, 0, stream>>>(xh, Wvh, bv, raw);
  v_transpose_kernel<<<(nSD + 255) / 256, 256, 0, stream>>>(raw, Vth);

  // SDPA (flash, WMMA, TDM-staged K/V)
  const int ATTN_GRID = HEADS * (S_LEN / 16) / 8;  // 384
  attn_kernel<<<ATTN_GRID, 256, 0, stream>>>(Qh, Kh, Vth, attn);

  // Output projection straight into d_out
  gemm_wmma_kernel<<<GEMM_GRID, 256, 0, stream>>>(attn, Woh, bo, out);
}
